// MFA_87067577025371
// MI455X (gfx1250) — matrix operand order repaired and checked by reading, not compiled, not run on hardware
//
#include <hip/hip_runtime.h>

// ---------------------------------------------------------------------------
// Non-local attention block, flash-style fused softmax for MI455X (gfx1250).
//   B=2, C=64, Ci=32, H=W=96, N=9216.
//   Pass1: k,v,q projections (v,q -> f16 for WMMA; q pre-scaled by log2(e)
//          so scores feed v_exp_f32 directly; k kept f32)
//   Pass2: Z[n] = sum_m exp(s[n,m]),  s = V^T Q   (WMMA 16x16x32 f16)
//   Pass3: k~ = k * 256 / Z
//   Pass4: av[c,m] = sum_n k~[c,n] * exp(s[n,m])  (recompute s, exp in-reg,
//          feed second WMMA; never materialize the 1.36 GB attention matrix)
//   Pass5: Wg + BN1 + Wo + BN2 + residual + relu  (undo the 256 scale)
// ---------------------------------------------------------------------------

#define NPOS 9216
#define NB 2
#define CI 32
#define CC 64
#define KSCALE 256.0f
#define LOG2E 1.4426950408889634f

typedef __attribute__((ext_vector_type(16))) _Float16 v16h;
typedef __attribute__((ext_vector_type(8)))  float    v8f;
typedef __attribute__((ext_vector_type(4)))  unsigned int u32x4;

union AB {
    v16h        v;
    u32x4       q[2];
    _Float16    h[16];
    unsigned int u[8];
};

union H8 {
    _Float16    h[8];
    unsigned int u[4];
};

// ---------------------------------------------------------------------------
// Kernel 1: 1x1 projections.  thread = (b, n).
// writes vT [B][N][32] f16, qT [B][N][32] f16 (q * log2e), kf [B][32][N] f32
// ---------------------------------------------------------------------------
__global__ __launch_bounds__(256) void proj_kernel(
    const float* __restrict__ xA, const float* __restrict__ xB,
    const float* __restrict__ Wk, const float* __restrict__ bk,
    const float* __restrict__ Wv, const float* __restrict__ bv,
    const float* __restrict__ Wq, const float* __restrict__ bq,
    _Float16* __restrict__ vT, _Float16* __restrict__ qT,
    float* __restrict__ kf)
{
    int t = blockIdx.x * blockDim.x + threadIdx.x;
    if (t >= NB * NPOS) return;
    int b = t / NPOS, n = t % NPOS;

    float x[CC];
    const float* xa = xA + (size_t)b * CC * NPOS + n;
#pragma unroll
    for (int c = 0; c < CC; ++c) x[c] = xa[(size_t)c * NPOS];

    for (int o = 0; o < CI; ++o) {
        float sk = bk[o], sv = bv[o];
        const float* wk = Wk + o * CC;
        const float* wv = Wv + o * CC;
#pragma unroll
        for (int c = 0; c < CC; ++c) { sk += wk[c] * x[c]; sv += wv[c] * x[c]; }
        kf[((size_t)(b * CI + o)) * NPOS + n] = sk;
        vT[((size_t)(b * NPOS + n)) * CI + o] = (_Float16)sv;
    }

    const float* xb = xB + (size_t)b * CC * NPOS + n;
#pragma unroll
    for (int c = 0; c < CC; ++c) x[c] = xb[(size_t)c * NPOS];

    for (int o = 0; o < CI; ++o) {
        float sq = bq[o];
        const float* wq = Wq + o * CC;
#pragma unroll
        for (int c = 0; c < CC; ++c) sq += wq[c] * x[c];
        // fold log2(e) into q: scores then feed v_exp_f32 (exp2) directly
        qT[((size_t)(b * NPOS + n)) * CI + o] = (_Float16)(sq * LOG2E);
    }
}

// ---------------------------------------------------------------------------
// Kernel 2: Z[n] = sum_m exp(s[n,m]).  One wave per (b, 16-row strip of n).
// ---------------------------------------------------------------------------
__global__ __launch_bounds__(256) void zsum_kernel(
    const _Float16* __restrict__ vT, const _Float16* __restrict__ qT,
    float* __restrict__ Z)
{
    const int lane = threadIdx.x & 31;
    const int wave = blockIdx.x * (blockDim.x >> 5) + (threadIdx.x >> 5);
    const int b  = wave / (NPOS / 16);
    const int n0 = (wave % (NPOS / 16)) * 16;
    const int hi   = (lane >= 16);
    const int col  = lane & 15;
    const int koff = hi ? 8 : 0;

    // A-operand: v rows n0..n0+15  (16-bit A layout: K chunks {0..7,16..23}/{8..15,24..31})
    AB a;
    {
        const _Float16* p = vT + ((size_t)(b * NPOS + n0 + col)) * CI + koff;
        a.q[0] = *(const u32x4*)(p);
        a.q[1] = *(const u32x4*)(p + 16);
    }

    float zacc[8];
#pragma unroll
    for (int j = 0; j < 8; ++j) zacc[j] = 0.0f;

    for (int mt = 0; mt < NPOS / 16; ++mt) {
        const int m0 = mt * 16;
        AB bq;
        const _Float16* p = qT + ((size_t)(b * NPOS + m0 + col)) * CI + koff;
        bq.q[0] = *(const u32x4*)(p);
        bq.q[1] = *(const u32x4*)(p + 16);

        v8f z = {};
        v8f s = __builtin_amdgcn_wmma_f32_16x16x32_f16(
            false, a.v, false, bq.v, (short)0, z, false, false);
#pragma unroll
        for (int j = 0; j < 8; ++j) zacc[j] += __builtin_amdgcn_exp2f(s[j]);
    }

    // reduce across the 16-lane half-wave (columns of this strip)
#pragma unroll
    for (int j = 0; j < 8; ++j) {
        zacc[j] += __shfl_xor(zacc[j], 1, 32);
        zacc[j] += __shfl_xor(zacc[j], 2, 32);
        zacc[j] += __shfl_xor(zacc[j], 4, 32);
        zacc[j] += __shfl_xor(zacc[j], 8, 32);
    }
    if (col == 0) {
        const int rbase = hi ? 8 : 0;  // lanes 0..15 hold rows 0..7; 16..31 rows 8..15
#pragma unroll
        for (int j = 0; j < 8; ++j)
            Z[(size_t)b * NPOS + n0 + rbase + j] = zacc[j];
    }
}

// ---------------------------------------------------------------------------
// Kernel 3: k~ = k * KSCALE / Z  (f16), layout [B][32][N]
// ---------------------------------------------------------------------------
__global__ __launch_bounds__(256) void kscale_kernel(
    const float* __restrict__ kf, const float* __restrict__ Z,
    _Float16* __restrict__ kh)
{
    int t = blockIdx.x * blockDim.x + threadIdx.x;
    if (t >= NB * NPOS) return;
    int b = t / NPOS, n = t % NPOS;
    float s = KSCALE / Z[(size_t)b * NPOS + n];
#pragma unroll
    for (int c = 0; c < CI; ++c) {
        size_t idx = ((size_t)(b * CI + c)) * NPOS + n;
        kh[idx] = (_Float16)(kf[idx] * s);
    }
}

// ---------------------------------------------------------------------------
// Kernel 4: main fused pass.  One wave per (b, 16-col m tile).
//   per 32-wide n chunk: 2 WMMA (scores) -> exp2 -> packed repack -> 2 WMMA.
// ---------------------------------------------------------------------------
__global__ __launch_bounds__(256) void attn_kernel(
    const _Float16* __restrict__ vT, const _Float16* __restrict__ qT,
    const _Float16* __restrict__ kh, float* __restrict__ av)
{
    const int lane = threadIdx.x & 31;
    const int wave = blockIdx.x * (blockDim.x >> 5) + (threadIdx.x >> 5);
    const int b  = wave / (NPOS / 16);
    const int m0 = (wave % (NPOS / 16)) * 16;
    const int hi   = (lane >= 16);
    const int col  = lane & 15;
    const int koff = hi ? 8 : 0;

    // B-operand of score WMMA: q columns m0..m0+15 (loop invariant)
    AB bq;
    {
        const _Float16* p = qT + ((size_t)(b * NPOS + m0 + col)) * CI + koff;
        bq.q[0] = *(const u32x4*)(p);
        bq.q[1] = *(const u32x4*)(p + 16);
    }

    v8f acc0 = {};  // channels 0..15
    v8f acc1 = {};  // channels 16..31

    const _Float16* vwave = vT + ((size_t)(b * NPOS + col)) * CI + koff;
    const _Float16* kwave = kh + ((size_t)(b * CI + col)) * NPOS + koff;

    for (int nc = 0; nc < NPOS / 32; ++nc) {
        const int n0 = nc * 32;

        // A tiles of V (rows n0..n0+15 and n0+16..n0+31)
        AB a0, a1;
        const _Float16* vb = vwave + (size_t)n0 * CI;
        a0.q[0] = *(const u32x4*)(vb);
        a0.q[1] = *(const u32x4*)(vb + 16);
        a1.q[0] = *(const u32x4*)(vb + 16 * CI);
        a1.q[1] = *(const u32x4*)(vb + 16 * CI + 16);
        __builtin_prefetch(vb + 32 * CI, 0, 1);

        v8f z = {};
        v8f s0 = __builtin_amdgcn_wmma_f32_16x16x32_f16(
            false, a0.v, false, bq.v, (short)0, z, false, false);
        v8f s1 = __builtin_amdgcn_wmma_f32_16x16x32_f16(
            false, a1.v, false, bq.v, (short)0, z, false, false);

        // exp2 each lane's own 8 values, convert+pack to f16 dwords FIRST,
        // then exchange packed dwords with lane^16 (8 ds ops, not 16).
        H8 p0, p1;
#pragma unroll
        for (int j = 0; j < 8; ++j) {
            p0.h[j] = (_Float16)__builtin_amdgcn_exp2f(s0[j]);
            p1.h[j] = (_Float16)__builtin_amdgcn_exp2f(s1[j]);
        }
        AB be;
#pragma unroll
        for (int j = 0; j < 4; ++j) {
            unsigned int x0 = __shfl_xor(p0.u[j], 16, 32);  // partner's packed S0 rows
            unsigned int x1 = __shfl_xor(p1.u[j], 16, 32);  // partner's packed S1 rows
            // halves 0..7 : own S0 rows (K=8h..8h+7)
            be.u[j]     = p0.u[j];
            // halves 8..15: own S1 rows (K=16+8h..16+8h+7)
            be.u[j + 4] = p1.u[j];
        }

        // A tiles of k~ (channel rows, K over n)
        AB k0, k1;
        const _Float16* kb = kwave + n0;
        k0.q[0] = *(const u32x4*)(kb);
        k0.q[1] = *(const u32x4*)(kb + 16);
        k1.q[0] = *(const u32x4*)(kb + (size_t)16 * NPOS);
        k1.q[1] = *(const u32x4*)(kb + (size_t)16 * NPOS + 16);
        __builtin_prefetch(kb + 32, 0, 1);

        acc0 = __builtin_amdgcn_wmma_f32_16x16x32_f16(
            false, k0.v, false, be.v, (short)0, acc0, false, false);
        acc1 = __builtin_amdgcn_wmma_f32_16x16x32_f16(
            false, k1.v, false, be.v, (short)0, acc1, false, false);
    }

    // store av [B][32][N] f32
#pragma unroll
    for (int r = 0; r < 8; ++r) {
        int c = r + (hi ? 8 : 0);
        av[((size_t)(b * CI + c)) * NPOS + m0 + col]      = acc0[r];
        av[((size_t)(b * CI + c + 16)) * NPOS + m0 + col] = acc1[r];
    }
}

// ---------------------------------------------------------------------------
// Kernel 5: epilogue.  thread = (b, m).
// ---------------------------------------------------------------------------
__global__ __launch_bounds__(256) void out_kernel(
    const float* __restrict__ av, const float* __restrict__ Wg,
    const float* __restrict__ g1g, const float* __restrict__ g1b,
    const float* __restrict__ g1m, const float* __restrict__ g1v,
    const float* __restrict__ Wo, const float* __restrict__ bo,
    const float* __restrict__ g2g, const float* __restrict__ g2b,
    const float* __restrict__ g2m, const float* __restrict__ g2v,
    const float* __restrict__ xB, float* __restrict__ out)
{
    int t = blockIdx.x * blockDim.x + threadIdx.x;
    if (t >= NB * NPOS) return;
    int b = t / NPOS, m = t % NPOS;

    float a[CI];
#pragma unroll
    for (int c = 0; c < CI; ++c)
        a[c] = av[((size_t)(b * CI + c)) * NPOS + m] * (1.0f / KSCALE);

    float t1[CI];
    for (int o = 0; o < CI; ++o) {
        float s = 0.0f;
        const float* wg = Wg + o * CI;
#pragma unroll
        for (int c = 0; c < CI; ++c) s += wg[c] * a[c];
        s = (s - g1m[o]) * (g1g[o] * rsqrtf(g1v[o] + 1e-5f)) + g1b[o];
        t1[o] = s;
    }

    for (int j = 0; j < CC; ++j) {
        float s = bo[j];
        const float* wo = Wo + j * CI;
#pragma unroll
        for (int o = 0; o < CI; ++o) s += wo[o] * t1[o];
        s = (s - g2m[j]) * (g2g[j] * rsqrtf(g2v[j] + 1e-5f)) + g2b[j];
        s += xB[((size_t)(b * CC + j)) * NPOS + m];
        out[((size_t)(b * CC + j)) * NPOS + m] = fmaxf(s, 0.0f);
    }
}

// ---------------------------------------------------------------------------
extern "C" void kernel_launch(void* const* d_in, const int* in_sizes, int n_in,
                              void* d_out, int out_size, void* d_ws, size_t ws_size,
                              hipStream_t stream)
{
    const float* xA = (const float*)d_in[0];
    const float* xB = (const float*)d_in[1];
    const float* Wk = (const float*)d_in[2];
    const float* bk = (const float*)d_in[3];
    const float* Wv = (const float*)d_in[4];
    const float* bv = (const float*)d_in[5];
    const float* Wq = (const float*)d_in[6];
    const float* bq = (const float*)d_in[7];
    const float* Wg = (const float*)d_in[8];
    const float* g1g = (const float*)d_in[9];
    const float* g1b = (const float*)d_in[10];
    const float* g1m = (const float*)d_in[11];
    const float* g1v = (const float*)d_in[12];
    const float* Wo = (const float*)d_in[13];
    const float* bo = (const float*)d_in[14];
    const float* g2g = (const float*)d_in[15];
    const float* g2b = (const float*)d_in[16];
    const float* g2m = (const float*)d_in[17];
    const float* g2v = (const float*)d_in[18];
    float* out = (float*)d_out;

    // workspace layout (total ~8.1 MB, 256B-aligned sections)
    char* ws = (char*)d_ws;
    _Float16* vT = (_Float16*)(ws + 0);        // 2*9216*32*2  = 1179648
    _Float16* qT = (_Float16*)(ws + 1179648);  //              = 1179648
    float*    kf = (float*)   (ws + 2359296);  // 2*32*9216*4  = 2359296
    _Float16* kh = (_Float16*)(ws + 4718592);  //              = 1179648
    float*    Zb = (float*)   (ws + 5898240);  // 2*9216*4     =   73728
    float*    av = (float*)   (ws + 5971968);  // 2*32*9216*4  = 2359296

    const int nthreads = NB * NPOS;            // 18432
    const int tb = 256;
    const int nblk = (nthreads + tb - 1) / tb; // 72
    const int nwaveblk = (NB * (NPOS / 16)) / (tb / 32); // 1152 waves / 8 = 144

    proj_kernel<<<nblk, tb, 0, stream>>>(xA, xB, Wk, bk, Wv, bv, Wq, bq, vT, qT, kf);
    zsum_kernel<<<nwaveblk, tb, 0, stream>>>(vT, qT, Zb);
    kscale_kernel<<<nblk, tb, 0, stream>>>(kf, Zb, kh);
    attn_kernel<<<nwaveblk, tb, 0, stream>>>(vT, qT, kh, av);
    out_kernel<<<nblk, tb, 0, stream>>>(av, Wg, g1g, g1b, g1m, g1v,
                                        Wo, bo, g2g, g2b, g2m, g2v, xB, out);
}
